// SageSparseLinearAttention_40269613367331
// MI455X (gfx1250) — 16-bit and 8-bit matrix operand order repaired and checked by reading, compiled, not run on hardware
//
#include <hip/hip_runtime.h>
#include <math.h>

typedef __attribute__((ext_vector_type(8)))  int    v8i;
typedef __attribute__((ext_vector_type(8)))  float  v8f;
typedef __attribute__((ext_vector_type(16))) __bf16 v16bf;

#define BB     2
#define HH     24
#define LL     4096
#define DD     64
#define BLKQ   128
#define BLKK   64
#define NQ     32
#define NK     64
#define TOPK   7
#define BH     (BB*HH)
#define SM_SCALE 0.125f
#define E4M3_DIV 199.1111111f

__device__ __forceinline__ float bf16r(float x) { return (float)(__bf16)x; }

__device__ __forceinline__ float e4m3q(float x) {
  float ax = fabsf(x);
  if (!(ax > 0.0f)) return 0.0f;
  if (ax > 448.0f) ax = 448.0f;
  int e; (void)frexpf(ax, &e);
  int ue = e - 1;
  float step = (ue < -6) ? ldexpf(1.0f, -9) : ldexpf(1.0f, ue - 3);
  float qv = rintf(ax / step) * step;
  if (qv > 448.0f) qv = 448.0f;
  return copysignf(qv, x);
}

__global__ void k_km_vscale(const float* __restrict__ k, const float* __restrict__ v,
                            float* __restrict__ km, float* __restrict__ vscale) {
  int bh = blockIdx.x, d = threadIdx.x;
  const float* kp = k + (size_t)bh * LL * DD + d;
  const float* vp = v + (size_t)bh * LL * DD + d;
  float s = 0.f, vm = 0.f;
  for (int l = 0; l < LL; ++l) {
    s  += bf16r(kp[(size_t)l * DD]);
    vm  = fmaxf(vm, fabsf(bf16r(vp[(size_t)l * DD])));
  }
  km[bh * DD + d]     = s * (1.0f / LL);
  vscale[bh * DD + d] = vm / E4M3_DIV + 1e-8f;
}

__global__ void k_pq_qscale(const float* __restrict__ q,
                            float* __restrict__ pq, float* __restrict__ qscale) {
  __shared__ float red[DD];
  int iq = blockIdx.x, bh = blockIdx.y, d = threadIdx.x;
  const float* qp = q + ((size_t)bh * LL + (size_t)iq * BLKQ) * DD + d;
  float s = 0.f, am = 0.f;
  for (int r = 0; r < BLKQ; ++r) {
    float x = qp[(size_t)r * DD];
    s += x;
    am = fmaxf(am, fabsf(bf16r(x)));
  }
  pq[((size_t)bh * NQ + iq) * DD + d] = s * (1.0f / BLKQ);
  red[d] = am;
  __syncthreads();
  for (int off = 32; off > 0; off >>= 1) {
    if (d < off) red[d] = fmaxf(red[d], red[d + off]);
    __syncthreads();
  }
  if (d == 0) qscale[(size_t)bh * NQ + iq] = red[0] / 127.0f + 1e-8f;
}

__global__ void k_pk_kscale(const float* __restrict__ k, const float* __restrict__ km,
                            float* __restrict__ pk, float* __restrict__ kscale) {
  __shared__ float red[DD];
  int ik = blockIdx.x, bh = blockIdx.y, d = threadIdx.x;
  const float* kp = k + ((size_t)bh * LL + (size_t)ik * BLKK) * DD + d;
  float kmd = km[bh * DD + d];
  float s = 0.f, am = 0.f;
  for (int r = 0; r < BLKK; ++r) {
    float x = kp[(size_t)r * DD];
    s += x;
    am = fmaxf(am, fabsf(bf16r(x) - kmd));
  }
  pk[((size_t)bh * NK + ik) * DD + d] = s * (1.0f / BLKK);
  red[d] = am;
  __syncthreads();
  for (int off = 32; off > 0; off >>= 1) {
    if (d < off) red[d] = fmaxf(red[d], red[d + off]);
    __syncthreads();
  }
  if (d == 0) kscale[(size_t)bh * NK + ik] = red[0] / 127.0f + 1e-8f;
}

__global__ void k_topk(const float* __restrict__ pq, const float* __restrict__ pk,
                       int* __restrict__ lut) {
  int bh = blockIdx.x, iq = threadIdx.x;
  float sc[NK];
  const float* pqv = pq + ((size_t)bh * NQ + iq) * DD;
  for (int j = 0; j < NK; ++j) {
    const float* pkv = pk + ((size_t)bh * NK + j) * DD;
    float dot = 0.f;
    for (int d = 0; d < DD; ++d) dot += pqv[d] * pkv[d];
    sc[j] = dot * SM_SCALE;
  }
  for (int t = 0; t < TOPK; ++t) {
    int best = 0; float bv = sc[0];
    for (int j = 1; j < NK; ++j) if (sc[j] > bv) { bv = sc[j]; best = j; }
    lut[((size_t)bh * NQ + iq) * TOPK + t] = best;
    sc[best] = -3.0e38f;
  }
}

__global__ __launch_bounds__(256) void k_sparse_attn(
    const float* __restrict__ q, const float* __restrict__ k, const float* __restrict__ v,
    const float* __restrict__ km, const float* __restrict__ vscale,
    const float* __restrict__ qscale, const float* __restrict__ kscale,
    const int* __restrict__ lut, float* __restrict__ out) {
  __shared__ signed char q8[BLKQ * DD];
  __shared__ signed char k8[BLKK * DD];
  __shared__ __bf16     vb[BLKK * DD];
  __shared__ __bf16     pst[8][16 * BLKK];
  __shared__ float kms[DD], vss[DD];

  int iq = blockIdx.x, bh = blockIdx.y;
  int tid = threadIdx.x, wave = tid >> 5, lane = tid & 31;
  int m = lane & 15, h = lane >> 4;

  if (tid < DD) { kms[tid] = km[bh * DD + tid]; vss[tid] = vscale[bh * DD + tid]; }
  float sq = qscale[bh * NQ + iq];

  size_t qbase = ((size_t)bh * LL + (size_t)iq * BLKQ) * DD;
  for (int i = tid; i < BLKQ * DD; i += 256) {
    float f = bf16r(q[qbase + i]);
    float r = fminf(fmaxf(rintf(f / sq), -127.f), 127.f);
    q8[i] = (signed char)(int)r;
  }
  __syncthreads();

  v8i Aq;
#pragma unroll
  for (int vv = 0; vv < 8; ++vv) {
    int kb = (vv >> 1) * 16 + h * 8 + (vv & 1) * 4;
    const signed char* row = &q8[(wave * 16 + m) * DD + kb];
    unsigned u = (unsigned)(unsigned char)row[0]
               | ((unsigned)(unsigned char)row[1] << 8)
               | ((unsigned)(unsigned char)row[2] << 16)
               | ((unsigned)(unsigned char)row[3] << 24);
    Aq[vv] = (int)u;
  }

  v8f Oacc[4];
  float mrun[8], lrun[8];
#pragma unroll
  for (int dt = 0; dt < 4; ++dt)
#pragma unroll
    for (int r = 0; r < 8; ++r) Oacc[dt][r] = 0.f;
#pragma unroll
  for (int r = 0; r < 8; ++r) { mrun[r] = -3.0e38f; lrun[r] = 0.f; }

  for (int t = 0; t < TOPK; ++t) {
    int blk = lut[((size_t)bh * NQ + iq) * TOPK + t];
    float skt = kscale[bh * NK + blk];
    size_t kbb = ((size_t)bh * LL + (size_t)blk * BLKK) * DD;

    for (int i = tid; i < BLKK * DD; i += 256) {
      int d = i & (DD - 1);
      float kf = bf16r(k[kbb + i]) - kms[d];
      float r = fminf(fmaxf(rintf(kf / skt), -127.f), 127.f);
      k8[i] = (signed char)(int)r;
      float vf = bf16r(v[kbb + i]) / vss[d];
      vb[i] = (__bf16)e4m3q(vf);
    }
    if (t + 1 < TOPK) {
      int nb = lut[((size_t)bh * NQ + iq) * TOPK + t + 1];
      size_t nbase = ((size_t)bh * LL + (size_t)nb * BLKK) * DD;
      __builtin_prefetch(&k[nbase + tid * 16], 0, 0);
      __builtin_prefetch(&v[nbase + tid * 16], 0, 0);
    }
    __syncthreads();

    float scl = sq * skt * SM_SCALE;
    float sarr[4][8];
#pragma unroll
    for (int jt = 0; jt < 4; ++jt) {
      v8i Bk;
#pragma unroll
      for (int vv = 0; vv < 8; ++vv) {
        int kb = (vv >> 1) * 16 + h * 8 + (vv & 1) * 4;
        const signed char* col = &k8[(jt * 16 + m) * DD + kb];
        unsigned u = (unsigned)(unsigned char)col[0]
                   | ((unsigned)(unsigned char)col[1] << 8)
                   | ((unsigned)(unsigned char)col[2] << 16)
                   | ((unsigned)(unsigned char)col[3] << 24);
        Bk[vv] = (int)u;
      }
      v8i c = {0, 0, 0, 0, 0, 0, 0, 0};
      c = __builtin_amdgcn_wmma_i32_16x16x64_iu8(true, Aq, true, Bk, c, false, false);
#pragma unroll
      for (int r = 0; r < 8; ++r) sarr[jt][r] = (float)c[r] * scl;
    }

    float mnew[8], alpha[8], rsum[8];
#pragma unroll
    for (int r = 0; r < 8; ++r) {
      float mx = fmaxf(fmaxf(sarr[0][r], sarr[1][r]), fmaxf(sarr[2][r], sarr[3][r]));
#pragma unroll
      for (int o = 8; o >= 1; o >>= 1) mx = fmaxf(mx, __shfl_xor(mx, o, 32));
      mnew[r] = fmaxf(mrun[r], mx);
      alpha[r] = expf(mrun[r] - mnew[r]);
      mrun[r] = mnew[r];
      rsum[r] = 0.f;
    }
#pragma unroll
    for (int jt = 0; jt < 4; ++jt) {
#pragma unroll
      for (int r = 0; r < 8; ++r) {
        float p = expf(sarr[jt][r] - mnew[r]);
        rsum[r] += p;
        pst[wave][(r + 8 * h) * BLKK + jt * 16 + m] = (__bf16)p;
      }
    }
#pragma unroll
    for (int r = 0; r < 8; ++r) {
      float s = rsum[r];
#pragma unroll
      for (int o = 8; o >= 1; o >>= 1) s += __shfl_xor(s, o, 32);
      lrun[r] = lrun[r] * alpha[r] + s;
    }
#pragma unroll
    for (int dt = 0; dt < 4; ++dt)
#pragma unroll
      for (int r = 0; r < 8; ++r) Oacc[dt][r] *= alpha[r];

#pragma unroll
    for (int c2 = 0; c2 < 2; ++c2) {
      v16bf Ap;
#pragma unroll
      for (int e = 0; e < 16; ++e) {
        int vv = e >> 1;
        int kk = h * 16 + e;
        Ap[e] = pst[wave][m * BLKK + c2 * 32 + kk];
      }
#pragma unroll
      for (int dt = 0; dt < 4; ++dt) {
        v16bf Bv;
#pragma unroll
        for (int e = 0; e < 16; ++e) {
          int kk = h * 16 + e;
          Bv[e] = vb[(c2 * 32 + kk) * DD + dt * 16 + m];
        }
        Oacc[dt] = __builtin_amdgcn_wmma_f32_16x16x32_bf16(false, Ap, false, Bv,
                                                           (short)0, Oacc[dt], false, false);
      }
    }
    __syncthreads();
  }

#pragma unroll
  for (int dt = 0; dt < 4; ++dt) {
#pragma unroll
    for (int r = 0; r < 8; ++r) {
      int d = dt * 16 + m;
      int row = iq * BLKQ + wave * 16 + r + 8 * h;
      out[((size_t)bh * LL + row) * DD + d] = Oacc[dt][r] / lrun[r] * vss[d];
    }
  }
}

__global__ void k_ck(const float* __restrict__ k, __bf16* __restrict__ ck) {
  size_t row = (size_t)blockIdx.x * blockDim.x + threadIdx.x;
  if (row >= (size_t)BH * LL) return;
  const float* kp = k + row * DD;
  float vals[DD]; float mx = -3.0e38f;
  for (int d = 0; d < DD; ++d) { vals[d] = bf16r(kp[d]); mx = fmaxf(mx, vals[d]); }
  float s = 0.f;
  for (int d = 0; d < DD; ++d) { vals[d] = expf(vals[d] - mx); s += vals[d]; }
  float inv = 1.f / s;
  for (int d = 0; d < DD; ++d) ck[row * DD + d] = (__bf16)(vals[d] * inv);
}

__global__ void k_ksum(const __bf16* __restrict__ ck, float* __restrict__ ksum) {
  int bh = blockIdx.x, d = threadIdx.x;
  const __bf16* p = ck + (size_t)bh * LL * DD + d;
  float s = 0.f;
  for (int l = 0; l < LL; ++l) s += (float)p[(size_t)l * DD];
  ksum[bh * DD + d] = s;
}

__global__ __launch_bounds__(512) void k_kvsum(const __bf16* __restrict__ ck,
                                               const float* __restrict__ v,
                                               float* __restrict__ kvsum) {
  int bh = blockIdx.x;
  int tid = threadIdx.x, wave = tid >> 5, lane = tid & 31;
  int m = lane & 15, h = lane >> 4;
  int dt = wave >> 2, et = wave & 3;
  const __bf16* ckb = ck + (size_t)bh * LL * DD;
  const float*  vbp = v  + (size_t)bh * LL * DD;
  v8f C = {0.f, 0.f, 0.f, 0.f, 0.f, 0.f, 0.f, 0.f};
  for (int l0 = 0; l0 < LL; l0 += 32) {
    v16bf A, Bv;
#pragma unroll
    for (int e = 0; e < 16; ++e) {
      int vv = e >> 1;
      int kk = h * 16 + e;
      A[e] = ckb[(size_t)(l0 + kk) * DD + dt * 16 + m];
      int kk2 = h * 16 + e;
      Bv[e] = (__bf16)vbp[(size_t)(l0 + kk2) * DD + et * 16 + m];
    }
    C = __builtin_amdgcn_wmma_f32_16x16x32_bf16(false, A, false, Bv, (short)0, C, false, false);
  }
#pragma unroll
  for (int r = 0; r < 8; ++r) {
    int d = dt * 16 + r + 8 * h;
    int e = et * 16 + m;
    kvsum[((size_t)bh * DD + d) * DD + e] = C[r];
  }
}

__global__ __launch_bounds__(128) void k_final(const float* __restrict__ q,
    const float* __restrict__ kvsum, const float* __restrict__ ksum,
    const float* __restrict__ W, const float* __restrict__ bproj,
    float* __restrict__ out) {
  __shared__ float  cq[16][DD];
  __shared__ float  den[16];
  __shared__ __bf16 ol[16][DD];
  int lt = blockIdx.x, bh = blockIdx.y;
  int l0 = lt * 16;
  int tid = threadIdx.x, wave = tid >> 5, lane = tid & 31;
  int m = lane & 15, h = lane >> 4;

  if (tid < 16) {
    const float* qp = q + ((size_t)bh * LL + l0 + tid) * DD;
    float vals[DD]; float mx = -3.0e38f;
    for (int d = 0; d < DD; ++d) { vals[d] = bf16r(qp[d]); mx = fmaxf(mx, vals[d]); }
    float s = 0.f;
    for (int d = 0; d < DD; ++d) { vals[d] = expf(vals[d] - mx); s += vals[d]; }
    float inv = 1.f / s, dn = 0.f;
    for (int d = 0; d < DD; ++d) {
      float c = vals[d] * inv;
      cq[tid][d] = c;
      dn += c * ksum[bh * DD + d];
    }
    den[tid] = dn;
  }
  __syncthreads();

  int et = wave;
  v8f C = {0.f, 0.f, 0.f, 0.f, 0.f, 0.f, 0.f, 0.f};
#pragma unroll
  for (int c2 = 0; c2 < 2; ++c2) {
    v16bf A, Bv;
#pragma unroll
    for (int e = 0; e < 16; ++e) {
      int vv = e >> 1;
      int kk = h * 16 + e;
      A[e] = (__bf16)cq[m][c2 * 32 + kk];
      int kk2 = h * 16 + e;
      Bv[e] = (__bf16)kvsum[((size_t)bh * DD + c2 * 32 + kk2) * DD + et * 16 + m];
    }
    C = __builtin_amdgcn_wmma_f32_16x16x32_bf16(false, A, false, Bv, (short)0, C, false, false);
  }
#pragma unroll
  for (int r = 0; r < 8; ++r) {
    int M = r + 8 * h;
    ol[M][et * 16 + m] = (__bf16)(C[r] / den[M]);
  }
  __syncthreads();

  v8f P = {0.f, 0.f, 0.f, 0.f, 0.f, 0.f, 0.f, 0.f};
#pragma unroll
  for (int c2 = 0; c2 < 2; ++c2) {
    v16bf A, Bv;
#pragma unroll
    for (int e = 0; e < 16; ++e) {
      int vv = e >> 1;
      int kk = h * 16 + e;
      A[e] = ol[m][c2 * 32 + kk];
      int kk2 = h * 16 + e;
      Bv[e] = (__bf16)W[(size_t)(et * 16 + m) * DD + c2 * 32 + kk2];
    }
    P = __builtin_amdgcn_wmma_f32_16x16x32_bf16(false, A, false, Bv, (short)0, P, false, false);
  }
#pragma unroll
  for (int r = 0; r < 8; ++r) {
    int M = r + 8 * h;
    int e = et * 16 + m;
    size_t idx = ((size_t)bh * LL + l0 + M) * DD + e;
    out[idx] += P[r] + bproj[e];
  }
}

extern "C" void kernel_launch(void* const* d_in, const int* in_sizes, int n_in,
                              void* d_out, int out_size, void* d_ws, size_t ws_size,
                              hipStream_t stream) {
  (void)in_sizes; (void)n_in; (void)out_size; (void)ws_size;
  const float* q  = (const float*)d_in[0];
  const float* k  = (const float*)d_in[1];
  const float* v  = (const float*)d_in[2];
  const float* W  = (const float*)d_in[3];
  const float* bp = (const float*)d_in[4];
  float* out = (float*)d_out;
  char* ws = (char*)d_ws;

  auto align256 = [](size_t x) { return (x + 255) & ~(size_t)255; };
  size_t off = 0;
  float* km     = (float*)(ws + off); off = align256(off + (size_t)BH * DD * 4);
  float* vscale = (float*)(ws + off); off = align256(off + (size_t)BH * DD * 4);
  float* qscale = (float*)(ws + off); off = align256(off + (size_t)BH * NQ * 4);
  float* kscale = (float*)(ws + off); off = align256(off + (size_t)BH * NK * 4);
  float* pq     = (float*)(ws + off); off = align256(off + (size_t)BH * NQ * DD * 4);
  float* pk     = (float*)(ws + off); off = align256(off + (size_t)BH * NK * DD * 4);
  int*   lut    = (int*)  (ws + off); off = align256(off + (size_t)BH * NQ * TOPK * 4);
  float* ksum   = (float*)(ws + off); off = align256(off + (size_t)BH * DD * 4);
  float* kvsum  = (float*)(ws + off); off = align256(off + (size_t)BH * DD * DD * 4);
  __bf16* ck    = (__bf16*)(ws + off); off = align256(off + (size_t)BH * LL * DD * 2);

  k_km_vscale<<<BH, DD, 0, stream>>>(k, v, km, vscale);
  k_pq_qscale<<<dim3(NQ, BH), DD, 0, stream>>>(q, pq, qscale);
  k_pk_kscale<<<dim3(NK, BH), DD, 0, stream>>>(k, km, pk, kscale);
  k_topk<<<BH, NQ, 0, stream>>>(pq, pk, lut);
  k_sparse_attn<<<dim3(NQ, BH), 256, 0, stream>>>(q, k, v, km, vscale, qscale, kscale, lut, out);
  k_ck<<<(BH * LL + 255) / 256, 256, 0, stream>>>(k, ck);
  k_ksum<<<BH, DD, 0, stream>>>(ck, ksum);
  k_kvsum<<<BH, 512, 0, stream>>>(ck, v, kvsum);
  k_final<<<dim3(LL / 16, BH), 128, 0, stream>>>(q, kvsum, ksum, W, bp, out);
}
